// CRF_66597762892245
// MI455X (gfx1250) — matrix operand order repaired and checked by reading, not compiled, not run on hardware
//
#include <hip/hip_runtime.h>
#include <hip/hip_bf16.h>
#include <float.h>

// ---------------------------------------------------------------------------
// Problem constants (from reference): B=32, T=2048, F=1024, U=512
// d_out layout (floats, concatenated in return order):
//   [0 .. 65535]                     viterbi_sequence (B*T)
//   [65536 .. 65536+33554431]        potentials (B*T*U)
//   [33619968 .. 33619999]           sequence_lengths (B)
//   [33620000 .. 33882143]           transitions copy (U*U)
// ---------------------------------------------------------------------------
#define BATCH 32
#define TLEN  2048
#define FDIM  1024
#define UDIM  512

typedef __bf16  bf16;
typedef __attribute__((ext_vector_type(16))) __bf16 v16bf;
typedef __attribute__((ext_vector_type(8)))  __bf16 bf16x8;
typedef __attribute__((ext_vector_type(8)))  float  v8f;
typedef __attribute__((ext_vector_type(4)))  float  f32x4;
typedef __attribute__((ext_vector_type(4)))  int    v4i;

// Async global->LDS path (ASYNCcnt), guarded so the file compiles on any
// toolchain; falls back to register staging if the builtins are absent.
// Probe-learned prototype (from hipcc diagnostic): param0 is a *global*
// (address_space(1)) int4*, param1 the LDS (address_space(3)) int4*.
#if defined(__gfx1250__) && \
    __has_builtin(__builtin_amdgcn_global_load_async_to_lds_b128) && \
    __has_builtin(__builtin_amdgcn_s_wait_asynccnt)
#define HAVE_ASYNC_LDS 1
#else
#define HAVE_ASYNC_LDS 0
#endif

// ---------------------------------------------------------------------------
// Kernel 1: potentials = x @ W + b  (+ boundary energies fused in epilogue)
// block = 256 threads (8 waves), tile 64(M) x 128(N), K-step 32.
// Waves arranged 2(M) x 4(N); each wave owns 32x32 = 2x2 WMMA 16x16 tiles.
// A (x) and B (W) staged to LDS as bf16; B staged transposed so each lane's
// WMMA B fragment is a contiguous 16-element K run (two 16B ds reads).
// ---------------------------------------------------------------------------
#define BK 32
#define LDA_PAD 40   // 32 + 8 bf16 pad
#define LDB_PAD 40

__global__ __launch_bounds__(256)
void crf_potentials_wmma(const float* __restrict__ X,   // (B*T, F)
                         const float* __restrict__ W,   // (F, U)
                         const float* __restrict__ bias,// (U)
                         const float* __restrict__ lb,  // (U)
                         const float* __restrict__ rb,  // (U)
                         float* __restrict__ out)       // (B*T, U)
{
    __shared__ bf16 sA[64 * LDA_PAD];        // 64 rows x 32 K (padded)
    __shared__ bf16 sB[128 * LDB_PAD];       // 128 cols x 32 K (transposed, padded)

    const int m0 = blockIdx.x * 64;
    const int n0 = blockIdx.y * 128;

    const int tid    = threadIdx.x;
    const int lane   = tid & 31;
    const int wave   = tid >> 5;
    const int wave_m = wave >> 2;            // 0..1
    const int wave_n = wave & 3;             // 0..3

    // staging coordinates
    const int ar = tid >> 2;                 // 0..63   A row
    const int ac = (tid & 3) * 8;            // 0,8,16,24
    const int bk = tid >> 3;                 // 0..31   W row (K)
    const int bn = (tid & 7) * 16;           // 0..112  W col base

    v8f acc[2][2];
    #pragma unroll
    for (int i = 0; i < 2; ++i)
        #pragma unroll
        for (int j = 0; j < 2; ++j)
            acc[i][j] = (v8f)0.0f;

    for (int k0 = 0; k0 < FDIM; k0 += BK) {
        __syncthreads();
        // ---- stage A tile (64x32 f32 -> bf16), coalesced float4 loads
        {
            const float* src = X + (size_t)(m0 + ar) * FDIM + k0 + ac;
            f32x4 v0 = *(const f32x4*)(src);
            f32x4 v1 = *(const f32x4*)(src + 4);
            bf16* dst = &sA[ar * LDA_PAD + ac];
            #pragma unroll
            for (int i = 0; i < 4; ++i) dst[i]     = (bf16)v0[i];
            #pragma unroll
            for (int i = 0; i < 4; ++i) dst[4 + i] = (bf16)v1[i];
        }
        // ---- stage B tile (32x128 f32 -> bf16, transposed into LDS)
        {
            const float* src = W + (size_t)(k0 + bk) * UDIM + n0 + bn;
            #pragma unroll
            for (int q = 0; q < 4; ++q) {
                f32x4 v = *(const f32x4*)(src + q * 4);
                #pragma unroll
                for (int i = 0; i < 4; ++i)
                    sB[(bn + q * 4 + i) * LDB_PAD + bk] = (bf16)v[i];
            }
        }
        __syncthreads();

        // ---- WMMA fragments
        // A: lane<16 -> K[0..15]; lane>=16 -> K[16..31]
        const int akoff = (lane >> 4) * 16;
        v16bf afrag[2];
        #pragma unroll
        for (int tm = 0; tm < 2; ++tm) {
            const int r = wave_m * 32 + tm * 16 + (lane & 15);
            bf16x8 lo = *(const bf16x8*)&sA[r * LDA_PAD + akoff];
            bf16x8 hi = *(const bf16x8*)&sA[r * LDA_PAD + akoff + 8];
            afrag[tm] = __builtin_shufflevector(lo, hi,
                0,1,2,3,4,5,6,7, 8,9,10,11,12,13,14,15);
        }
        // B: lane<16 -> K[0..15]; lane>=16 -> K[16..31] (contiguous in sB)
        const int bkoff = (lane >> 4) * 16;
        v16bf bfrag[2];
        #pragma unroll
        for (int tn = 0; tn < 2; ++tn) {
            const int n = wave_n * 32 + tn * 16 + (lane & 15);
            bf16x8 lo = *(const bf16x8*)&sB[n * LDB_PAD + bkoff];
            bf16x8 hi = *(const bf16x8*)&sB[n * LDB_PAD + bkoff + 8];
            bfrag[tn] = __builtin_shufflevector(lo, hi,
                0,1,2,3,4,5,6,7, 8,9,10,11,12,13,14,15);
        }

        #pragma unroll
        for (int tm = 0; tm < 2; ++tm)
            #pragma unroll
            for (int tn = 0; tn < 2; ++tn)
                acc[tm][tn] = __builtin_amdgcn_wmma_f32_16x16x32_bf16(
                    false, afrag[tm], false, bfrag[tn],
                    (short)0, acc[tm][tn], false, false);
    }

    // ---- epilogue: + bias; boundary energies handled by *block-uniform*
    // scalar branches (a 64-row tile contains t==0 iff m0%T==0, and
    // t==T-1 iff m0%T==T-64), so the hot path has no divergence.
    const bool has_first = ((m0 & (TLEN - 1)) == 0);
    const bool has_last  = ((m0 & (TLEN - 1)) == (TLEN - 64));
    const int  row_half  = (lane >> 4) * 8;  // C layout: lanes>=16 -> M += 8
    #pragma unroll
    for (int tm = 0; tm < 2; ++tm) {
        #pragma unroll
        for (int tn = 0; tn < 2; ++tn) {
            const int col = n0 + wave_n * 32 + tn * 16 + (lane & 15);
            const float bcol = bias[col];
            #pragma unroll
            for (int r = 0; r < 8; ++r) {
                const int row_local = wave_m * 32 + tm * 16 + r + row_half;
                float v = acc[tm][tn][r] + bcol;
                if (has_first && row_local == 0)  v += lb[col];
                if (has_last  && row_local == 63) v += rb[col];
                out[(size_t)(m0 + row_local) * UDIM + col] = v;
            }
        }
    }
}

// ---------------------------------------------------------------------------
// Kernel 2: Viterbi forward. One block per batch, 512 threads (1 per tag u).
// The 1MB transition matrix is streamed through LDS in 128KB chunks,
// double-buffered (2x128KB ping-pong fits CDNA5's 320KB WGP LDS), using
// async global->LDS copies (ASYNCcnt) so the copy of chunk c+1 overlaps the
// max-plus VALU work on chunk c. state[512] lives in LDS; chunk reads are
// bank-conflict-free (consecutive u -> consecutive banks).
// ---------------------------------------------------------------------------
#define CH          64                         // rows (v) per chunk
#define NCHUNK      (UDIM / CH)                // 8
#define CHUNK_ELEMS (CH * UDIM)                // 32768 floats = 128KB

__device__ __forceinline__
void stage_chunk(float* dstLds, const float* srcGlb, int tid)
{
    // 128KB / 16B = 8192 units; 512 threads -> 16 units each, coalesced.
#if HAVE_ASYNC_LDS
    #pragma unroll
    for (int j = 0; j < 16; ++j) {
        const int unit = j * 512 + tid;
        __builtin_amdgcn_global_load_async_to_lds_b128(
            (__attribute__((address_space(1))) v4i*)(srcGlb + unit * 4),
            (__attribute__((address_space(3))) v4i*)(dstLds + unit * 4),
            0, 0);
    }
#else
    #pragma unroll
    for (int j = 0; j < 16; ++j) {
        const int unit = j * 512 + tid;
        *(f32x4*)(dstLds + unit * 4) = *(const f32x4*)(srcGlb + unit * 4);
    }
#endif
}

__device__ __forceinline__ void stage_wait_and_barrier()
{
#if HAVE_ASYNC_LDS
    __builtin_amdgcn_s_wait_asynccnt(0);
#endif
    __syncthreads();
}

__global__ __launch_bounds__(512)
void crf_viterbi_fwd(const float* __restrict__ pot,    // (B,T,U) in d_out
                     const float* __restrict__ trans,  // (U,U)
                     unsigned short* __restrict__ bp,  // (B, T-1, U)
                     int* __restrict__ last_tag)       // (B)
{
    extern __shared__ __align__(16) float buf[];       // 2 * CHUNK_ELEMS
    __shared__ __align__(16) float state[UDIM];
    __shared__ float rv[UDIM];
    __shared__ int   ri[UDIM];

    const int b   = blockIdx.x;
    const int u   = threadIdx.x;
    const int tid = threadIdx.x;

    state[u] = pot[((size_t)b * TLEN) * UDIM + u];

    // preload chunk 0 into buffer 0
    stage_chunk(buf, trans, tid);
    stage_wait_and_barrier();

    int cur = 0;
    for (int t = 1; t < TLEN; ++t) {
        __builtin_prefetch(&pot[((size_t)b * TLEN + t) * UDIM + u], 0, 1);

        float best = -FLT_MAX;
        int   bidx = 0;

        #pragma unroll 1
        for (int c = 0; c < NCHUNK; ++c) {
            const int nxt = cur ^ 1;
            const int cn  = (c + 1) & (NCHUNK - 1);    // trans is constant:
            stage_chunk(buf + nxt * CHUNK_ELEMS,       // chunk stream is
                        trans + (size_t)cn * CHUNK_ELEMS, tid); // periodic

            const float* bufc = buf + cur * CHUNK_ELEMS;
            #pragma unroll 4
            for (int v = 0; v < CH; v += 4) {
                f32x4 s4 = *(const f32x4*)&state[c * CH + v];
                #pragma unroll
                for (int j = 0; j < 4; ++j) {
                    float cand = s4[j] + bufc[(v + j) * UDIM + u];
                    if (cand > best) { best = cand; bidx = c * CH + v + j; }
                }
            }
            stage_wait_and_barrier();
            cur = nxt;
        }

        best += pot[((size_t)b * TLEN + t) * UDIM + u];
        bp[((size_t)b * (TLEN - 1) + (t - 1)) * UDIM + u] = (unsigned short)bidx;

        // chunk-7 barrier guaranteed all state reads done; publish new state
        state[u] = best;
        __syncthreads();
    }

    // argmax of final state (prefer lowest index on ties, like jnp.argmax)
    rv[u] = state[u];
    ri[u] = u;
    __syncthreads();
    for (int s = UDIM / 2; s > 0; s >>= 1) {
        if (u < s) {
            if (rv[u + s] > rv[u] || (rv[u + s] == rv[u] && ri[u + s] < ri[u])) {
                rv[u] = rv[u + s];
                ri[u] = ri[u + s];
            }
        }
        __syncthreads();
    }
    if (u == 0) last_tag[b] = ri[0];
}

// ---------------------------------------------------------------------------
// Kernel 3: backtrace. One wave, one lane per batch (serial in t by nature).
// ---------------------------------------------------------------------------
__global__ __launch_bounds__(32)
void crf_backtrace(const unsigned short* __restrict__ bp,
                   const int* __restrict__ last_tag,
                   float* __restrict__ seq)             // (B,T) as float
{
    const int b = threadIdx.x;
    if (b >= BATCH) return;
    int tag = last_tag[b];
    seq[(size_t)b * TLEN + (TLEN - 1)] = (float)tag;
    for (int t = TLEN - 2; t >= 0; --t) {
        tag = (int)bp[((size_t)b * (TLEN - 1) + t) * UDIM + tag];
        seq[(size_t)b * TLEN + t] = (float)tag;
    }
}

// ---------------------------------------------------------------------------
// Kernel 4: tail outputs — sequence_lengths (=T) and transitions pass-through.
// ---------------------------------------------------------------------------
__global__ void crf_tail(const float* __restrict__ trans,
                         float* __restrict__ lens,
                         float* __restrict__ trans_out)
{
    const int i = blockIdx.x * blockDim.x + threadIdx.x;
    if (i < UDIM * UDIM) trans_out[i] = trans[i];
    if (i < BATCH)       lens[i] = (float)TLEN;
}

// ---------------------------------------------------------------------------
extern "C" void kernel_launch(void* const* d_in, const int* in_sizes, int n_in,
                              void* d_out, int out_size, void* d_ws, size_t ws_size,
                              hipStream_t stream)
{
    const float* x     = (const float*)d_in[0];  // (B,T,F)
    const float* W     = (const float*)d_in[1];  // (F,U)
    const float* bias  = (const float*)d_in[2];  // (U)
    const float* trans = (const float*)d_in[3];  // (U,U)
    const float* lb    = (const float*)d_in[4];  // (U)
    const float* rb    = (const float*)d_in[5];  // (U)

    float* out      = (float*)d_out;
    float* seq_out  = out;                                   // B*T
    float* pot      = out + (size_t)BATCH * TLEN;            // B*T*U
    float* lens     = pot + (size_t)BATCH * TLEN * UDIM;     // B
    float* trans_o  = lens + BATCH;                          // U*U

    const size_t bp_bytes = (size_t)BATCH * (TLEN - 1) * UDIM * sizeof(unsigned short);
    unsigned short* bp = (unsigned short*)d_ws;
    int* last_tag      = (int*)((char*)d_ws + bp_bytes);

    // 1) GEMM + bias + boundary energies (WMMA bf16 -> f32)
    dim3 ggrid((BATCH * TLEN) / 64, UDIM / 128);
    crf_potentials_wmma<<<ggrid, 256, 0, stream>>>(x, W, bias, lb, rb, pot);

    // 2) tail outputs (independent of GEMM)
    crf_tail<<<(UDIM * UDIM + 255) / 256, 256, 0, stream>>>(trans, lens, trans_o);

    // 3) Viterbi forward (reads potentials region of d_out);
    //    dynamic LDS = 2 x 128KB transition ping-pong (CDNA5 320KB WGP LDS)
    crf_viterbi_fwd<<<BATCH, UDIM, 2 * CHUNK_ELEMS * sizeof(float), stream>>>(
        pot, trans, bp, last_tag);

    // 4) backtrace -> sequence
    crf_backtrace<<<1, 32, 0, stream>>>(bp, last_tag, seq_out);
}
